// HypersphereBlock_90452011254069
// MI455X (gfx1250) — matrix operand order repaired and checked by reading, not compiled, not run on hardware
//
#include <hip/hip_runtime.h>
#include <math.h>

// ---------------------------------------------------------------------------
// CDNA5 (gfx1250) nGPT-style transformer block.
// All GEMMs + attention run on v_wmma_f32_16x16x32_bf16 (wave32).
// GEMM operand tiles are staged global->LDS with the Tensor Data Mover
// (tensor_load_to_lds, TENSORcnt), double-buffered, with TDM padding
// reproducing the bank-conflict-free 80B LDS row stride.
// ---------------------------------------------------------------------------

typedef __bf16 bf16;
typedef unsigned int u32;
typedef unsigned long long u64;
typedef __attribute__((ext_vector_type(16))) __bf16 v16bf;
typedef __attribute__((ext_vector_type(8)))  float  v8f;
typedef __attribute__((ext_vector_type(4)))  unsigned int u32x4;
typedef __attribute__((ext_vector_type(8)))  int i32x8;
typedef __attribute__((ext_vector_type(4)))  int i32x4;

union FragU { v16bf v; uint4 q[2]; };

static __device__ inline v8f wmma_bf16(v16bf a, v16bf b, v8f c) {
  // 8 args: (neg_a, A, neg_b, B, c_mod, C, reuse_a, reuse_b)
  return __builtin_amdgcn_wmma_f32_16x16x32_bf16(
      false, a, false, b, (short)0, c, false, false);
}

// A-matrix fragment, 16x32 bf16 (rows row0..row0+15, K slice kb..kb+31),
// row-major source with leading dim `ld` (elements).
// ISA layout (16-bit A 16x32): lane m<16 holds K = kb+{0..7, 16..23},
// lane m+16 holds K = kb+{8..15, 24..31}, row = lane&15.
static __device__ inline v16bf load_frag_a(const bf16* __restrict__ src,
                                           int ld, int row0, int kb, int lane) {
  const int row = row0 + (lane & 15);
  const int k0  = kb + ((lane & 16) ? 8 : 0);
  const bf16* p = src + (size_t)row * ld + k0;
  FragU f;
  f.q[0] = *(const uint4*)(p);        // 8 bf16
  f.q[1] = *(const uint4*)(p + 16);   // 8 bf16, +16 elements
  return f.v;
}

// B-matrix fragment, 32x16 bf16 for C = A * W^T:  B[k][n] = W[n0+n][kb+k].
// ISA layout: lane n<16 holds column n, K = kb+{0..7, 16..23};
// lane n+16 holds column n, K = kb+{8..15, 24..31}.
static __device__ inline v16bf load_frag_b(const bf16* __restrict__ src,
                                           int ld, int n0, int kb, int lane) {
  const int col = n0 + (lane & 15);
  const int k0  = kb + ((lane & 16) ? 8 : 0);
  const bf16* p = src + (size_t)col * ld + k0;
  FragU f;
  f.q[0] = *(const uint4*)(p);        // 8 bf16
  f.q[1] = *(const uint4*)(p + 16);   // 8 bf16, +16 elements
  return f.v;
}

static __device__ inline v8f zero_v8f() {
  return (v8f){0.f, 0.f, 0.f, 0.f, 0.f, 0.f, 0.f, 0.f};
}

// ---------------------------------------------------------------------------
// Tensor Data Mover: async DMA of a 2D bf16 tile (tile_d0 x tile_d1 elements)
// from a row-major tensor (leading-dim stride_elems) into LDS at lds_off.
// Padding: +4 DWORDs every 16 DWORDs (64B row -> 80B LDS stride, LDT=40).
// D# bitfields per CDNA5 ISA 8.3/8.4. 2-D tensor -> groups 2/3 unused (zero).
// ---------------------------------------------------------------------------
static __device__ inline void tdm_load_2d(u32 lds_off, const bf16* gptr,
                                          u32 tensor_d0, u32 tensor_d1,
                                          u32 tile_d0, u32 tile_d1,
                                          u32 stride_elems) {
  const u64 ga = (u64)(uintptr_t)gptr;
  u32x4 g0;
  g0[0] = 1u;                                        // count=1 (valid D#)
  g0[1] = lds_off;                                   // lds_addr (bytes)
  g0[2] = (u32)(ga & 0xFFFFFFFFu);                   // global_addr[95:64]
  g0[3] = (u32)((ga >> 32) & 0x01FFFFFFu) | (2u << 30);  // addr hi | type=2
  i32x8 g1;
  g1[0] = (int)((1u << 16) |                         // data_size=1 (2 bytes)
                (1u << 20) |                         // pad_enable
                (3u << 22) |                         // pad_interval: 16 DWORDs
                (3u << 25));                         // pad_amount: 4 DWORDs
  g1[1] = (int)((tensor_d0 & 0xFFFFu) << 16);        // dim0[15:0] (atomic=0)
  g1[2] = (int)((tensor_d0 >> 16) | ((tensor_d1 & 0xFFFFu) << 16));
  g1[3] = (int)((tensor_d1 >> 16) | (tile_d0 << 16));
  g1[4] = (int)(tile_d1 & 0xFFFFu);                  // tile_dim1 (tile_dim2=0)
  g1[5] = (int)stride_elems;                         // tensor_dim0_stride lo32
  g1[6] = 0;
  g1[7] = 0;
  const i32x4 z4 = {0, 0, 0, 0};
#if defined(__clang_major__) && (__clang_major__ >= 23)
  const i32x8 z8 = {0, 0, 0, 0, 0, 0, 0, 0};
  __builtin_amdgcn_tensor_load_to_lds(g0, g1, z4, z4, z8, 0);
#else
  __builtin_amdgcn_tensor_load_to_lds(g0, g1, z4, z4, 0);
#endif
}

// ---------------------------------------------------------------------------
// Weight cast f32 -> bf16 (grid-stride).
// ---------------------------------------------------------------------------
__global__ __launch_bounds__(256) void cast_bf16_kernel(
    const float* __restrict__ in, bf16* __restrict__ out, size_t n) {
  size_t i = (size_t)blockIdx.x * blockDim.x + threadIdx.x;
  const size_t stride = (size_t)gridDim.x * blockDim.x;
  for (; i < n; i += stride) out[i] = (bf16)in[i];
}

// ---------------------------------------------------------------------------
// Hypersphere norm: out[row] = x[row] / (||x[row]|| + eps) * sqrt(D) * g
// One 256-thread block per row, bf16 output.
// ---------------------------------------------------------------------------
__global__ __launch_bounds__(256) void hnorm_kernel(
    const float* __restrict__ x, const float* __restrict__ g,
    bf16* __restrict__ out, int D) {
  const int t = threadIdx.x;
  const float* row = x + (size_t)blockIdx.x * D;
  float ss = 0.f;
  for (int i = t; i < D; i += 256) { float v = row[i]; ss += v * v; }
  __shared__ float red[256];
  red[t] = ss;
  __syncthreads();
  for (int o = 128; o > 0; o >>= 1) {
    if (t < o) red[t] += red[t + o];
    __syncthreads();
  }
  const float scale = sqrtf((float)D) / (sqrtf(red[0]) + 1e-6f);
  bf16* orow = out + (size_t)blockIdx.x * D;
  for (int i = t; i < D; i += 256) orow[i] = (bf16)(row[i] * scale * g[i]);
}

// ---------------------------------------------------------------------------
// NT GEMM: C[M,N] = A[M,K] * W[N,K]^T, bf16 inputs, f32 accumulation, WMMA.
// Block tile 128x128, 8 waves (2 in M x 4 in N), wave tile 64x32 = 4x2 frags.
// K staged 32 at a time via double-buffered TDM loads (wave 0 -> A tile,
// wave 1 -> W tile; TDM ignores EXEC so branches are wave-uniform). TDM pad
// gives 40-element LDS rows -> conflict-free ds_load_b128 fragment gathers
// (20*r mod 64 bank rotation).
// MODE 0: store f32. MODE 1: store f32 + residual. MODE 2: GELU -> bf16.
// ---------------------------------------------------------------------------
template <int MODE>
__global__ __launch_bounds__(256) void gemm_nt_kernel(
    const bf16* __restrict__ A, const bf16* __restrict__ W,
    const float* __restrict__ res, float* __restrict__ outf,
    bf16* __restrict__ outb, int M, int N, int K) {
  constexpr int LDT = 40;  // padded LDS row stride in elements (80 bytes)
  __shared__ __align__(16) bf16 sA[2][128 * LDT];
  __shared__ __align__(16) bf16 sW[2][128 * LDT];

  const int tid  = threadIdx.x;
  const int lane = tid & 31;
  const int wave = tid >> 5;
  const int bm   = blockIdx.y * 128;
  const int bn   = blockIdx.x * 128;
  const int wm   = (wave >> 2) * 64;  // wave row offset inside tile
  const int wn   = (wave & 3) * 32;   // wave col offset inside tile

  v8f acc[4][2];
#pragma unroll
  for (int i = 0; i < 4; ++i)
#pragma unroll
    for (int j = 0; j < 2; ++j) acc[i][j] = zero_v8f();

  const u32 ldsA0 = (u32)(uintptr_t)&sA[0][0];
  const u32 ldsA1 = (u32)(uintptr_t)&sA[1][0];
  const u32 ldsW0 = (u32)(uintptr_t)&sW[0][0];
  const u32 ldsW1 = (u32)(uintptr_t)&sW[1][0];

  auto issue_tiles = [&](int kt, int buf) {
    const int kb = kt * 32;
    if (wave == 0) {
      tdm_load_2d(buf ? ldsA1 : ldsA0, A + (size_t)bm * K + kb,
                  (u32)K, (u32)M, 32u, 128u, (u32)K);
    } else if (wave == 1) {
      tdm_load_2d(buf ? ldsW1 : ldsW0, W + (size_t)bn * K + kb,
                  (u32)K, (u32)N, 32u, 128u, (u32)K);
    }
  };

  const int nk = K / 32;
  issue_tiles(0, 0);
  if (wave < 2) __builtin_amdgcn_s_wait_tensorcnt(0);
  __syncthreads();

  for (int kt = 0; kt < nk; ++kt) {
    const int cur = kt & 1;
    if (kt + 1 < nk) {
      issue_tiles(kt + 1, cur ^ 1);  // async DMA overlaps WMMA below
      // gfx1250 prefetch of the tile after next (global_prefetch_b8)
      if (kt + 2 < nk)
        __builtin_prefetch(W + (size_t)(bn + (tid >> 2)) * K + (kt + 2) * 32, 0, 1);
    }
    v16bf af[4], wf[2];
#pragma unroll
    for (int i = 0; i < 4; ++i)
      af[i] = load_frag_a(&sA[cur][0], LDT, wm + 16 * i, 0, lane);
#pragma unroll
    for (int j = 0; j < 2; ++j)
      wf[j] = load_frag_b(&sW[cur][0], LDT, wn + 16 * j, 0, lane);
#pragma unroll
    for (int i = 0; i < 4; ++i)
#pragma unroll
      for (int j = 0; j < 2; ++j) acc[i][j] = wmma_bf16(af[i], wf[j], acc[i][j]);
    if (kt + 1 < nk) {
      if (wave < 2) __builtin_amdgcn_s_wait_tensorcnt(0);
      __syncthreads();  // next tile visible; prev buffer free for next DMA
    }
  }

  // C layout: VGPR r, lanes<16 -> row r, lanes>=16 -> row r+8; col = lane&15.
  const int rbase = (lane & 16) ? 8 : 0;
  const int cn    = lane & 15;
#pragma unroll
  for (int i = 0; i < 4; ++i) {
#pragma unroll
    for (int j = 0; j < 2; ++j) {
#pragma unroll
      for (int r = 0; r < 8; ++r) {
        const int row = bm + wm + 16 * i + r + rbase;
        const int col = bn + wn + 16 * j + cn;
        const size_t off = (size_t)row * N + col;
        const float v = acc[i][j][r];
        if (MODE == 1) {
          outf[off] = v + res[off];
        } else if (MODE == 2) {
          outb[off] = (bf16)(0.5f * v * (1.0f + erff(v * 0.70710678118f)));
        } else {
          outf[off] = v;
        }
      }
    }
  }
}

// ---------------------------------------------------------------------------
// QK-norm + repack: q,k f32 [B,S,D] -> unit-normalized bf16 [B,H,S,hd].
// qk_norm_factor is folded into Q here so the flash inner loop needs no scale.
// One 128-thread block per (b,s,h) head vector (hd == 128).
// ---------------------------------------------------------------------------
__global__ __launch_bounds__(128) void qk_post_kernel(
    const float* __restrict__ qf, const float* __restrict__ kf,
    bf16* __restrict__ qo, bf16* __restrict__ ko,
    const float* __restrict__ scale_ptr, int S, int H, int hd) {
  const int t = threadIdx.x;
  const int h = blockIdx.x % H;
  const int s = (blockIdx.x / H) % S;
  const int b = blockIdx.x / (H * S);
  const size_t in_off  = ((size_t)b * S + s) * (size_t)(H * hd) + (size_t)h * hd + t;
  const size_t out_off = (((size_t)b * H + h) * S + s) * (size_t)hd + t;
  const float qv = qf[in_off];
  const float kv = kf[in_off];
  __shared__ float rq[128], rk[128];
  rq[t] = qv * qv;
  rk[t] = kv * kv;
  __syncthreads();
  for (int o = 64; o > 0; o >>= 1) {
    if (t < o) { rq[t] += rq[t + o]; rk[t] += rk[t + o]; }
    __syncthreads();
  }
  const float qs = scale_ptr[0] / (sqrtf(rq[0]) + 1e-6f);
  const float ks = 1.0f / (sqrtf(rk[0]) + 1e-6f);
  qo[out_off] = (bf16)(qv * qs);
  ko[out_off] = (bf16)(kv * ks);
}

// ---------------------------------------------------------------------------
// V repack: f32 [B,S,H*hd] -> bf16 V^T [B,H,hd,S] (coalesced writes).
// Transposed so flash-attention P*V B-fragments read contiguous memory.
// ---------------------------------------------------------------------------
__global__ __launch_bounds__(256) void v_post_kernel(
    const float* __restrict__ vf, bf16* __restrict__ vt,
    int Bn, int S, int H, int hd) {
  const size_t total = (size_t)Bn * H * hd * S;
  size_t i = (size_t)blockIdx.x * blockDim.x + threadIdx.x;
  const size_t stride = (size_t)gridDim.x * blockDim.x;
  for (; i < total; i += stride) {
    const int s = (int)(i % S);
    size_t r = i / S;
    const int d = (int)(r % hd); r /= hd;
    const int h = (int)(r % H);
    const int b = (int)(r / H);
    vt[i] = (bf16)vf[((size_t)b * S + s) * (size_t)(H * hd) + (size_t)h * hd + d];
  }
}

// ---------------------------------------------------------------------------
// Flash attention, causal, hd = 128, all-WMMA (scale pre-folded into Q).
// grid = (S/64, B*H); 128 threads = 4 waves, 16 q-rows per wave.
// Q kept resident in A-fragments; online softmax with half-wave shuffles;
// P repacked C-layout -> A-layout through a wave-private LDS tile
// (same-wave DS ordering guarantees store->load visibility, ISA 7.3).
// ---------------------------------------------------------------------------
__global__ __launch_bounds__(128) void flash_attn_kernel(
    const bf16* __restrict__ Q, const bf16* __restrict__ Kmat,
    const bf16* __restrict__ VT, bf16* __restrict__ O, int S, int H) {
  constexpr int HD = 128;
  const int lane = threadIdx.x & 31;
  const int wave = threadIdx.x >> 5;  // 0..3
  const int bh   = blockIdx.y;
  const int b    = bh / H;
  const int h    = bh % H;
  const int q0   = blockIdx.x * 64 + wave * 16;

  const bf16* Qh = Q    + (size_t)bh * S * HD;  // [S, hd]
  const bf16* Kh = Kmat + (size_t)bh * S * HD;  // [S, hd]
  const bf16* Vh = VT   + (size_t)bh * HD * S;  // [hd, S]

  v16bf qf[4];
#pragma unroll
  for (int t = 0; t < 4; ++t) qf[t] = load_frag_a(Qh, HD, q0, 32 * t, lane);

  v8f o[8];
  float m[8], l[8];
#pragma unroll
  for (int n = 0; n < 8; ++n) o[n] = zero_v8f();
#pragma unroll
  for (int r = 0; r < 8; ++r) { m[r] = -3.0e38f; l[r] = 0.f; }

  __shared__ __align__(16) bf16 psh[4][16 * 32];  // per-wave P tile, 1 KB each
  bf16* pw = &psh[wave][0];

  const int rbase = (lane & 16) ? 8 : 0;
  const int cn    = lane & 15;

  for (int kb = 0; kb < q0 + 16; kb += 32) {
    // S-tile = Q[16,128] x K[kb..kb+31,128]^T  (two 16-col WMMA tiles)
    v8f sc0 = zero_v8f(), sc1 = zero_v8f();
#pragma unroll
    for (int t = 0; t < 4; ++t) {
      sc0 = wmma_bf16(qf[t], load_frag_b(Kh, HD, kb,      32 * t, lane), sc0);
      sc1 = wmma_bf16(qf[t], load_frag_b(Kh, HD, kb + 16, 32 * t, lane), sc1);
    }
#pragma unroll
    for (int r = 0; r < 8; ++r) {
      const int row = q0 + r + rbase;
      float v0 = sc0[r];
      float v1 = sc1[r];
      if (kb + cn > row)      v0 = -3.0e38f;  // causal mask
      if (kb + 16 + cn > row) v1 = -3.0e38f;
      float tv = fmaxf(v0, v1);
#pragma unroll
      for (int off = 8; off >= 1; off >>= 1)   // stays within 16-lane N-group
        tv = fmaxf(tv, __shfl_xor(tv, off, 32));
      const float mnew = fmaxf(m[r], tv);
      const float corr = __expf(m[r] - mnew);
      const float p0   = __expf(v0 - mnew);
      const float p1   = __expf(v1 - mnew);
      float ps = p0 + p1;
#pragma unroll
      for (int off = 8; off >= 1; off >>= 1) ps += __shfl_xor(ps, off, 32);
      l[r] = l[r] * corr + ps;
      m[r] = mnew;
#pragma unroll
      for (int n = 0; n < 8; ++n) o[n][r] *= corr;
      const int rowl = r + rbase;
      pw[rowl * 32 + cn]      = (bf16)p0;
      pw[rowl * 32 + 16 + cn] = (bf16)p1;
    }
    // P (16x32) as A-fragment from LDS, then O += P * V
    v16bf pf = load_frag_a(pw, 32, 0, 0, lane);
#pragma unroll
    for (int n = 0; n < 8; ++n)
      o[n] = wmma_bf16(pf, load_frag_b(Vh, S, 16 * n, kb, lane), o[n]);
  }

  // Epilogue: merge heads, bf16 out [B,S,H*hd]
#pragma unroll
  for (int r = 0; r < 8; ++r) {
    const float inv = 1.0f / l[r];
    const int row = q0 + r + rbase;
    bf16* orow = O + ((size_t)b * S + row) * (size_t)(H * HD) + (size_t)h * HD;
#pragma unroll
    for (int n = 0; n < 8; ++n) orow[16 * n + cn] = (bf16)(o[n][r] * inv);
  }
}

// ---------------------------------------------------------------------------
// Host-side orchestration.
// ---------------------------------------------------------------------------
extern "C" void kernel_launch(void* const* d_in, const int* in_sizes, int n_in,
                              void* d_out, int out_size, void* d_ws, size_t ws_size,
                              hipStream_t stream) {
  (void)in_sizes; (void)n_in; (void)out_size; (void)ws_size;
  constexpr int Bc = 2, S = 2048, D = 2048, F = 8192, H = 16, HD = 128;
  constexpr int M = Bc * S;  // 4096 token rows

  const float* hs    = (const float*)d_in[0];
  const float* Wq    = (const float*)d_in[1];
  const float* Wk    = (const float*)d_in[2];
  const float* Wv    = (const float*)d_in[3];
  const float* Wo    = (const float*)d_in[4];
  const float* Win   = (const float*)d_in[5];
  const float* Wout  = (const float*)d_in[6];
  const float* qknf  = (const float*)d_in[7];  // 1-element device scalar
  const float* gattn = (const float*)d_in[8];
  const float* gmlp  = (const float*)d_in[9];

  char* ws = (char*)d_ws;
  const size_t MB = 1024ull * 1024ull;
  // Layout (272 MB peak, with reuse):
  bf16*  wq_b   = (bf16*)(ws +   0 * MB);  //  8 MB
  bf16*  wk_b   = (bf16*)(ws +   8 * MB);  //  8 MB
  bf16*  wv_b   = (bf16*)(ws +  16 * MB);  //  8 MB
  bf16*  wo_b   = (bf16*)(ws +  24 * MB);  //  8 MB
  bf16*  win_b  = (bf16*)(ws +  32 * MB);  // 32 MB
  bf16*  wout_b = (bf16*)(ws +  64 * MB);  // 32 MB
  bf16*  xn     = (bf16*)(ws +  96 * MB);  // 16 MB (norm out; reused for MLP in)
  float* qf32   = (float*)(ws + 112 * MB); // 32 MB
  float* kf32   = (float*)(ws + 144 * MB); // 32 MB
  float* vf32   = (float*)(ws + 176 * MB); // 32 MB
  bf16*  qb     = (bf16*)(ws + 208 * MB);  // 16 MB [B,H,S,hd]
  bf16*  kbh    = (bf16*)(ws + 224 * MB);  // 16 MB [B,H,S,hd]
  bf16*  vtb    = (bf16*)(ws + 240 * MB);  // 16 MB [B,H,hd,S]
  bf16*  attnb  = (bf16*)(ws + 256 * MB);  // 16 MB [B,S,D]
  float* hbuf   = (float*)(ws + 112 * MB); // 32 MB, reuses qf32 (dead by then)
  bf16*  ubuf   = (bf16*)(ws + 144 * MB);  // 64 MB, reuses kf32+vf32

  // 1) cast weights to bf16
  cast_bf16_kernel<<<512, 256, 0, stream>>>(Wq,   wq_b,   (size_t)D * D);
  cast_bf16_kernel<<<512, 256, 0, stream>>>(Wk,   wk_b,   (size_t)D * D);
  cast_bf16_kernel<<<512, 256, 0, stream>>>(Wv,   wv_b,   (size_t)D * D);
  cast_bf16_kernel<<<512, 256, 0, stream>>>(Wo,   wo_b,   (size_t)D * D);
  cast_bf16_kernel<<<512, 256, 0, stream>>>(Win,  win_b,  (size_t)F * D);
  cast_bf16_kernel<<<512, 256, 0, stream>>>(Wout, wout_b, (size_t)D * F);

  // 2) pre-attention hypersphere norm
  hnorm_kernel<<<M, 256, 0, stream>>>(hs, gattn, xn, D);

  // 3) QKV projections (WMMA GEMMs, TDM-staged)
  dim3 gd(D / 128, M / 128);
  gemm_nt_kernel<0><<<gd, 256, 0, stream>>>(xn, wq_b, nullptr, qf32, nullptr, M, D, D);
  gemm_nt_kernel<0><<<gd, 256, 0, stream>>>(xn, wk_b, nullptr, kf32, nullptr, M, D, D);
  gemm_nt_kernel<0><<<gd, 256, 0, stream>>>(xn, wv_b, nullptr, vf32, nullptr, M, D, D);

  // 4) QK-norm (+ fold qk_norm_factor into Q) + head-major repack; V transpose
  qk_post_kernel<<<Bc * S * H, 128, 0, stream>>>(qf32, kf32, qb, kbh, qknf, S, H, HD);
  v_post_kernel<<<2048, 256, 0, stream>>>(vf32, vtb, Bc, S, H, HD);

  // 5) causal flash attention (WMMA)
  dim3 ga(S / 64, Bc * H);
  flash_attn_kernel<<<ga, 128, 0, stream>>>(qb, kbh, vtb, attnb, S, H);

  // 6) output projection + residual -> h
  gemm_nt_kernel<1><<<gd, 256, 0, stream>>>(attnb, wo_b, hs, hbuf, nullptr, M, D, D);

  // 7) pre-MLP hypersphere norm
  hnorm_kernel<<<M, 256, 0, stream>>>(hbuf, gmlp, xn, D);

  // 8) MLP up-projection + exact GELU -> bf16
  dim3 gu(F / 128, M / 128);
  gemm_nt_kernel<2><<<gu, 256, 0, stream>>>(xn, win_b, nullptr, nullptr, ubuf, M, F, D);

  // 9) MLP down-projection + residual -> final f32 output
  gemm_nt_kernel<1><<<gd, 256, 0, stream>>>(ubuf, wout_b, hbuf, (float*)d_out, nullptr, M, D, F);
}
